// MultiHeadedAttention_60550448939172
// MI455X (gfx1250) — matrix operand order repaired and checked by reading, not compiled, not run on hardware
//
#include <hip/hip_runtime.h>
#include <hip/hip_bf16.h>
#include <math.h>

// ---------------- problem constants ----------------
constexpr int Bsz = 8, S = 512, D = 1024, H = 16, DH = 64;

// ---------------- WMMA types ----------------
typedef __attribute__((ext_vector_type(16))) _Float16 v16h;
typedef __attribute__((ext_vector_type(8)))  _Float16 v8h_t;   // 16 bytes
typedef __attribute__((ext_vector_type(8)))  float    v8f;
typedef __attribute__((ext_vector_type(4)))  unsigned int uint4v;
typedef __attribute__((ext_vector_type(8)))  int      int8v;
typedef __attribute__((ext_vector_type(4)))  int      int4v;

union Frag16 {
    v16h    v;
    v8h_t   h[2];
    _Float16 e[16];
};

static __device__ __forceinline__ v8f v8f_zero() {
    v8f z = {0.f,0.f,0.f,0.f,0.f,0.f,0.f,0.f};
    return z;
}

static __device__ __forceinline__ v8f wmma_f16(const Frag16& a, const Frag16& b, v8f c) {
    return __builtin_amdgcn_wmma_f32_16x16x32_f16(
        /*neg_a=*/false, a.v, /*neg_b=*/false, b.v,
        /*c_mod=*/(short)0, c, /*reuse_a=*/false, /*reuse_b=*/false);
}

// ---------------- Tensor Data Mover (CDNA5 TDM) ----------------
#if defined(__has_builtin)
#  if __has_builtin(__builtin_amdgcn_tensor_load_to_lds)
#    define USE_TDM 1
#  endif
#endif
#ifndef USE_TDM
#  define USE_TDM 0
#endif

#if USE_TDM
// DMA one 128-row x 128-byte contiguous f16 tile into LDS, inserting 16B of
// LDS padding after every 128B row (pad_interval = 32 DWORDs, pad_amount = 4
// DWORDs) so the landed layout has a 144B row stride (bank-conflict free for
// the 16-lane B-fragment reads). Issued by one wave; tracked by TENSORcnt.
static __device__ __forceinline__ void tdm_load_k_tile(const _Float16* gsrc, unsigned lds_off) {
    const unsigned long long ga = (unsigned long long)(uintptr_t)gsrc;
    uint4v g0;
    g0[0] = 1u;                                  // count=1 (valid user descriptor)
    g0[1] = lds_off;                             // D#.lds_addr (bytes)
    g0[2] = (unsigned)ga;                        // D#.global_addr[31:0]
    g0[3] = (unsigned)(ga >> 32) | (2u << 30);   // global_addr[56:32] | type=2 ("image")

    int8v g1;
    g1[0] = (int)((3u << 16)      // data_size = 8 bytes
                | (1u << 20)      // pad_enable
                | (4u << 22)      // pad_interval = 32 DWORDs (one 128B row)
                | (3u << 25));    // pad_amount  = 4 DWORDs (16B)
    g1[1] = (int)(16u << 16);     // tensor_dim0 = 16 elements (one row)
    g1[2] = (int)(128u << 16);    // tensor_dim0 hi=0 | tensor_dim1 lo = 128 rows
    g1[3] = (int)(16u << 16);     // tensor_dim1 hi=0 | tile_dim0 = 16
    g1[4] = (int)(128u);          // tile_dim1 = 128 | tile_dim2 = 0 (2D tile)
    g1[5] = (int)(16u);           // tensor_dim0_stride = 16 elements (rows contiguous)
    g1[6] = 0;
    g1[7] = 0;

    int4v gz4 = {0,0,0,0};
    int8v gz8 = {0,0,0,0,0,0,0,0};
    // 6-arg (clang-23 / therock-10.0) form: (g0, g1, g2, g3, extra_g, cpol)
    __builtin_amdgcn_tensor_load_to_lds(g0, g1, gz4, gz4, gz8, 0);
}
#endif

// =====================================================================
// K1: projection GEMM  out[m,n] = sum_k X[m,k] * W[n,k] + bias[n]
//     M = B*S = 4096, N = D = 1024, K = D = 1024.
//     Output f16 in (B,H,S,DH) layout (transposed==0) or (B,H,DH,S) (==1).
//     Block = 256 threads = 8 waves; 128x128 tile; K step 32; LDS double
//     buffered so next tile's global loads overlap current tile's WMMAs.
//     __launch_bounds__(256,1): keep acc + staging regs resident (no spills).
// =====================================================================
__global__ void __launch_bounds__(256, 1)
proj_kernel(const float* __restrict__ X,
            const float* __restrict__ W,
            const float* __restrict__ bias,
            _Float16* __restrict__ out,
            int transposed) {
    const int mb = blockIdx.x;      // 0..31  (M tiles of 128)
    const int nb = blockIdx.y;      // 0..7   (N tiles of 128)
    const int wave = threadIdx.x >> 5;
    const int lane = threadIdx.x & 31;
    const int lg   = lane & 15;     // lane within 16-group
    const int hi   = lane >> 4;     // which half of the wave

    __shared__ __align__(16) _Float16 As[2][128][48];  // 96B rows (16B aligned)
    __shared__ __align__(16) _Float16 Bs[2][128][48];

    // staging coordinates for this thread (4 float4 per matrix per tile)
    int srow[4], scg[4];
    #pragma unroll
    for (int rep = 0; rep < 4; ++rep) {
        const int li = threadIdx.x + 256 * rep;   // 0..1023
        srow[rep] = li >> 3;
        scg[rep]  = (li & 7) * 4;
    }

    float4 fa[4], fb[4];
    auto load_regs = [&](int kb) {
        #pragma unroll
        for (int rep = 0; rep < 4; ++rep) {
            fa[rep] = *(const float4*)(X + (size_t)(mb * 128 + srow[rep]) * D + kb + scg[rep]);
            fb[rep] = *(const float4*)(W + (size_t)(nb * 128 + srow[rep]) * D + kb + scg[rep]);
        }
    };
    auto store_regs = [&](int buf) {
        #pragma unroll
        for (int rep = 0; rep < 4; ++rep) {
            const int row = srow[rep], cg = scg[rep];
            As[buf][row][cg+0] = (_Float16)fa[rep].x;  As[buf][row][cg+1] = (_Float16)fa[rep].y;
            As[buf][row][cg+2] = (_Float16)fa[rep].z;  As[buf][row][cg+3] = (_Float16)fa[rep].w;
            Bs[buf][row][cg+0] = (_Float16)fb[rep].x;  Bs[buf][row][cg+1] = (_Float16)fb[rep].y;
            Bs[buf][row][cg+2] = (_Float16)fb[rep].z;  Bs[buf][row][cg+3] = (_Float16)fb[rep].w;
        }
    };

    v8f acc[8];
    #pragma unroll
    for (int nt = 0; nt < 8; ++nt) acc[nt] = v8f_zero();

    load_regs(0);
    store_regs(0);
    __syncthreads();

    for (int kb = 0, it = 0; kb < D; kb += 32, ++it) {
        const int cur   = it & 1;
        const bool more = (kb + 32 < D);
        if (more) {
            load_regs(kb + 32);                   // issue next-tile global loads early
            __builtin_prefetch(X + (size_t)(mb * 128 + srow[0]) * D + kb + 64, 0, 3);
        }

        // ---- A fragment: 16x32, M = lg, K-half selected by hi ----
        Frag16 a;
        const _Float16* ap = &As[cur][wave * 16 + lg][hi * 8];
        a.h[0] = *(const v8h_t*)(ap);        // K 0..7   (+8 for hi)
        a.h[1] = *(const v8h_t*)(ap + 16);   // K 16..23 (+8 for hi)

        #pragma unroll
        for (int nt = 0; nt < 8; ++nt) {
            // B fragment: 32x16, N = lg, K 0..7,16..23 (+8 for hi)
            Frag16 b;
            const _Float16* bp = &Bs[cur][nt * 16 + lg][hi * 8];
            b.h[0] = *(const v8h_t*)(bp);
            b.h[1] = *(const v8h_t*)(bp + 16);
            acc[nt] = wmma_f16(a, b, acc[nt]);
        }

        if (more) store_regs(cur ^ 1);
        __syncthreads();
    }

    // ---- epilogue: C layout lane=N=lg, VGPR r -> M = r + hi*8 ----
    #pragma unroll
    for (int nt = 0; nt < 8; ++nt) {
        const int ng = nb * 128 + nt * 16 + lg;     // output feature
        const int hh = ng >> 6, dh = ng & 63;
        const float bv = bias[ng];
        #pragma unroll
        for (int r = 0; r < 8; ++r) {
            const int mg = mb * 128 + wave * 16 + r + hi * 8;  // token index
            const int b = mg >> 9, s = mg & 511;
            const float val = acc[nt][r] + bv;
            const size_t idx = transposed
                ? ((size_t)((b * H + hh) * DH + dh)) * S + s
                : ((size_t)((b * H + hh) * S  + s )) * DH + dh;
            out[idx] = (_Float16)val;
        }
    }
}

// =====================================================================
// K2: per-(b,i) softmax rows for rel_attn and time_attn (h-independent;
//     R1 is constant along the softmax axis so it cancels exactly).
// =====================================================================
__global__ void rel_time_kernel(const float* __restrict__ rel,
                                const float* __restrict__ ts,
                                float* __restrict__ rel_sm,
                                float* __restrict__ time_sm) {
    const int bi = blockIdx.x;          // b*S + i
    const int i  = bi & (S - 1);
    const int j  = threadIdx.x;
    const int wid = threadIdx.x >> 5, lane = threadIdx.x & 31;
    __shared__ float red[16];
    const size_t idx = (size_t)bi * S + j;

    // ---------- rel softmax ----------
    {
        const float rv = rel[idx];
        float x = (rv == 0.0f) ? -10000.0f : rv;
        float m = x;
        #pragma unroll
        for (int d = 1; d < 32; d <<= 1) m = fmaxf(m, __shfl_xor(m, d, 32));
        if (lane == 0) red[wid] = m;
        __syncthreads();
        float bm = red[0];
        #pragma unroll
        for (int w = 1; w < 16; ++w) bm = fmaxf(bm, red[w]);
        __syncthreads();
        const float e = __expf(x - bm);
        float sum = e;
        #pragma unroll
        for (int d = 1; d < 32; d <<= 1) sum += __shfl_xor(sum, d, 32);
        if (lane == 0) red[wid] = sum;
        __syncthreads();
        float bs = 0.f;
        #pragma unroll
        for (int w = 0; w < 16; ++w) bs += red[w];
        __syncthreads();
        rel_sm[idx] = e / bs;
    }
    // ---------- time softmax ----------
    {
        const float x = (j > i) ? -1.0e30f : __expf(-fabsf(ts[idx]));
        float m = x;
        #pragma unroll
        for (int d = 1; d < 32; d <<= 1) m = fmaxf(m, __shfl_xor(m, d, 32));
        if (lane == 0) red[wid] = m;
        __syncthreads();
        float bm = red[0];
        #pragma unroll
        for (int w = 1; w < 16; ++w) bm = fmaxf(bm, red[w]);
        __syncthreads();
        const float e = __expf(x - bm);
        float sum = e;
        #pragma unroll
        for (int d = 1; d < 32; d <<= 1) sum += __shfl_xor(sum, d, 32);
        if (lane == 0) red[wid] = sum;
        __syncthreads();
        float bs = 0.f;
        #pragma unroll
        for (int w = 0; w < 16; ++w) bs += red[w];
        time_sm[idx] = e / bs;
    }
}

// =====================================================================
// K3a: scores = q k^T / 8, causal-masked; flash-style online row max/sum.
//      Key chunks are DMA'd into LDS by the Tensor Data Mover (double
//      buffered, TENSORcnt-synchronized); raw masked scores spill into the
//      prob output region. grid = (4, B*H); block = 256 (8 waves).
// =====================================================================
__global__ void __launch_bounds__(256, 1)
attn_scores_kernel(const _Float16* __restrict__ q16,
                   const _Float16* __restrict__ k16,
                   float* __restrict__ prob,
                   float* __restrict__ rowmax,
                   float* __restrict__ rowsum) {
    const int bh   = blockIdx.y;
    const int mblk = blockIdx.x;
    const int wave = threadIdx.x >> 5;
    const int lane = threadIdx.x & 31;
    const int lg   = lane & 15;
    const int hi   = lane >> 4;

    __shared__ __align__(16) _Float16 Ks[2][128][72];   // 144B rows (TDM-padded)

    // A fragments for this wave's 16 query rows (K = DH = 64 -> two 32-steps)
    const int irow = mblk * 128 + wave * 16 + lg;
    const _Float16* qrow = q16 + ((size_t)bh * S + irow) * DH;
    Frag16 a0, a1;
    {
        const int kb = hi * 8;
        a0.h[0] = *(const v8h_t*)(qrow + kb);
        a0.h[1] = *(const v8h_t*)(qrow + kb + 16);
        a1.h[0] = *(const v8h_t*)(qrow + 32 + kb);
        a1.h[1] = *(const v8h_t*)(qrow + 32 + kb + 16);
    }

    float m_run[8], l_run[8];
    #pragma unroll
    for (int r = 0; r < 8; ++r) { m_run[r] = -3.0e38f; l_run[r] = 0.f; }

#if USE_TDM
    if (wave == 0)
        tdm_load_k_tile(k16 + ((size_t)bh * S) * DH,
                        (unsigned)(uintptr_t)&Ks[0][0][0]);
#endif

    for (int nc = 0; nc < 4; ++nc) {
        const int cur = nc & 1;
#if USE_TDM
        if (wave == 0) __builtin_amdgcn_s_wait_tensorcnt(0);   // chunk nc landed
        __syncthreads();
        if (nc + 1 < 4 && wave == 0)                            // prefetch next chunk
            tdm_load_k_tile(k16 + ((size_t)bh * S + (nc + 1) * 128) * DH,
                            (unsigned)(uintptr_t)&Ks[cur ^ 1][0][0]);
#else
        {   // manual double-buffered staging fallback
            const int row  = threadIdx.x >> 1;
            const int part = (threadIdx.x & 1) * 32;
            const v8h_t* src = (const v8h_t*)(k16 + ((size_t)bh * S + nc * 128 + row) * DH + part);
            v8h_t* dst = (v8h_t*)&Ks[cur][row][part];
            dst[0] = src[0]; dst[1] = src[1]; dst[2] = src[2]; dst[3] = src[3];
        }
        __syncthreads();
#endif

        v8f acc[8];
        #pragma unroll
        for (int nt = 0; nt < 8; ++nt) {
            acc[nt] = v8f_zero();
            Frag16 b0, b1;
            const _Float16* kp = &Ks[cur][nt * 16 + lg][hi * 8];
            b0.h[0] = *(const v8h_t*)(kp);
            b0.h[1] = *(const v8h_t*)(kp + 16);
            b1.h[0] = *(const v8h_t*)(kp + 32);
            b1.h[1] = *(const v8h_t*)(kp + 48);
            acc[nt] = wmma_f16(a0, b0, acc[nt]);
            acc[nt] = wmma_f16(a1, b1, acc[nt]);
        }

        // scale + causal mask + online stats
        const float scale = 0.125f;   // 1/sqrt(64)
        #pragma unroll
        for (int r = 0; r < 8; ++r) {
            const int i = mblk * 128 + wave * 16 + r + hi * 8;
            float mx = -3.0e38f;
            #pragma unroll
            for (int nt = 0; nt < 8; ++nt) {
                const int j = nc * 128 + nt * 16 + lg;
                float s = acc[nt][r] * scale;
                if (j > i) s = -1.0e9f;
                acc[nt][r] = s;
                mx = fmaxf(mx, s);
            }
            #pragma unroll
            for (int d = 1; d < 16; d <<= 1) mx = fmaxf(mx, __shfl_xor(mx, d, 32));
            const float mnew = fmaxf(m_run[r], mx);
            float ssum = 0.f;
            #pragma unroll
            for (int nt = 0; nt < 8; ++nt) ssum += __expf(acc[nt][r] - mnew);
            #pragma unroll
            for (int d = 1; d < 16; d <<= 1) ssum += __shfl_xor(ssum, d, 32);
            l_run[r] = l_run[r] * __expf(m_run[r] - mnew) + ssum;
            m_run[r] = mnew;
        }

        // spill raw masked scores into prob region (normalized in K3b)
        #pragma unroll
        for (int nt = 0; nt < 8; ++nt) {
            const int j = nc * 128 + nt * 16 + lg;
            #pragma unroll
            for (int r = 0; r < 8; ++r) {
                const int i = mblk * 128 + wave * 16 + r + hi * 8;
                prob[((size_t)bh * S + i) * S + j] = acc[nt][r];
            }
        }
    }

    if (lg == 0) {
        #pragma unroll
        for (int r = 0; r < 8; ++r) {
            const int i = mblk * 128 + wave * 16 + r + hi * 8;
            rowmax[bh * S + i] = m_run[r];
            rowsum[bh * S + i] = l_run[r];
        }
    }
}

// =====================================================================
// K3b: normalize scores + mix with time/rel attention (in place, float4).
// =====================================================================
__global__ void combine_kernel(float* __restrict__ prob,
                               const float* __restrict__ rowmax,
                               const float* __restrict__ rowsum,
                               const float* __restrict__ rel_sm,
                               const float* __restrict__ time_sm,
                               const float* __restrict__ l1p,
                               const float* __restrict__ l2p) {
    const float l1 = *l1p, l2 = *l2p;
    const float ca = (1.f - l1) * (1.f - l2);
    const float ct = (1.f - l1) * l2;
    const size_t total4 = (size_t)Bsz * H * S * S / 4;
    for (size_t q = (size_t)blockIdx.x * blockDim.x + threadIdx.x;
         q < total4; q += (size_t)gridDim.x * blockDim.x) {
        const size_t idx = q * 4;
        const int j  = (int)(idx & (S - 1));
        const int i  = (int)((idx >> 9) & (S - 1));
        const int bh = (int)(idx >> 18);
        const int b  = bh >> 4;
        const float rm = rowmax[bh * S + i];
        const float ri = 1.f / rowsum[bh * S + i];
        const size_t bij = ((size_t)(b * S + i)) * S + j;
        const float4 s = *(const float4*)(prob + idx);
        const float4 t = *(const float4*)(time_sm + bij);
        const float4 r = *(const float4*)(rel_sm + bij);
        float4 o;
        o.x = ca * __expf(s.x - rm) * ri + ct * t.x + l1 * r.x;
        o.y = ca * __expf(s.y - rm) * ri + ct * t.y + l1 * r.y;
        o.z = ca * __expf(s.z - rm) * ri + ct * t.z + l1 * r.z;
        o.w = ca * __expf(s.w - rm) * ri + ct * t.w + l1 * r.w;
        *(float4*)(prob + idx) = o;
    }
}

// =====================================================================
// K4: out[b,h] = prob (512x512, f32 -> f16 on the fly) @ V^T-laid V.
//     grid = (4, B*H); wave owns 16 rows x full DH (4 n-subtiles); K = 512.
// =====================================================================
__global__ void __launch_bounds__(256, 1)
out_gemm_kernel(const float* __restrict__ prob,
                const _Float16* __restrict__ vT,
                float* __restrict__ out) {
    const int bh   = blockIdx.y;
    const int mblk = blockIdx.x;
    const int wave = threadIdx.x >> 5;
    const int lane = threadIdx.x & 31;
    const int lg   = lane & 15;
    const int hi   = lane >> 4;

    const int i = mblk * 128 + wave * 16 + lg;      // A row (M = lg)
    const float* prow = prob + ((size_t)bh * S + i) * S;

    v8f acc[4];
    #pragma unroll
    for (int nt = 0; nt < 4; ++nt) acc[nt] = v8f_zero();

    for (int ks = 0; ks < S; ks += 32) {
        Frag16 a;
        const int kb = ks + hi * 8;
        const float4 f0 = *(const float4*)(prow + kb);
        const float4 f1 = *(const float4*)(prow + kb + 4);
        const float4 f2 = *(const float4*)(prow + kb + 16);
        const float4 f3 = *(const float4*)(prow + kb + 20);
        a.e[0] = (_Float16)f0.x;  a.e[1] = (_Float16)f0.y;
        a.e[2] = (_Float16)f0.z;  a.e[3] = (_Float16)f0.w;
        a.e[4] = (_Float16)f1.x;  a.e[5] = (_Float16)f1.y;
        a.e[6] = (_Float16)f1.z;  a.e[7] = (_Float16)f1.w;
        a.e[8]  = (_Float16)f2.x; a.e[9]  = (_Float16)f2.y;
        a.e[10] = (_Float16)f2.z; a.e[11] = (_Float16)f2.w;
        a.e[12] = (_Float16)f3.x; a.e[13] = (_Float16)f3.y;
        a.e[14] = (_Float16)f3.z; a.e[15] = (_Float16)f3.w;

        #pragma unroll
        for (int nt = 0; nt < 4; ++nt) {
            Frag16 b;
            const _Float16* vp = vT + ((size_t)bh * DH + nt * 16 + lg) * S + ks + hi * 8;
            b.h[0] = *(const v8h_t*)(vp);
            b.h[1] = *(const v8h_t*)(vp + 16);
            acc[nt] = wmma_f16(a, b, acc[nt]);
        }
    }

    const int b = bh >> 4, hh = bh & 15;
    #pragma unroll
    for (int nt = 0; nt < 4; ++nt) {
        const int dcol = hh * DH + nt * 16 + lg;
        #pragma unroll
        for (int r = 0; r < 8; ++r) {
            const int s = mblk * 128 + wave * 16 + r + hi * 8;
            out[((size_t)(b * S + s)) * D + dcol] = acc[nt][r];
        }
    }
}

// =====================================================================
// launch
// =====================================================================
extern "C" void kernel_launch(void* const* d_in, const int* in_sizes, int n_in,
                              void* d_out, int out_size, void* d_ws, size_t ws_size,
                              hipStream_t stream) {
    (void)in_sizes; (void)n_in; (void)out_size; (void)ws_size;

    const float* query = (const float*)d_in[0];
    const float* key_  = (const float*)d_in[1];
    const float* value = (const float*)d_in[2];
    const float* rel   = (const float*)d_in[3];
    const float* ts    = (const float*)d_in[4];
    const float* l1    = (const float*)d_in[5];
    const float* l2    = (const float*)d_in[6];
    // d_in[7] = mask (causal triu, applied analytically), d_in[8] = encode_pos (unused)
    const float* Wq = (const float*)d_in[9];
    const float* bq = (const float*)d_in[10];
    const float* Wk = (const float*)d_in[11];
    const float* bk = (const float*)d_in[12];
    const float* Wv = (const float*)d_in[13];
    const float* bv = (const float*)d_in[14];
    // d_in[15..18] = Wc, bc, Wg, bg: R1 is constant along the softmax axis -> cancels.

    char* ws = (char*)d_ws;
    _Float16* q16    = (_Float16*)(ws);                 //  8 MB
    _Float16* k16    = (_Float16*)(ws + (8u  << 20));   //  8 MB
    _Float16* vT16   = (_Float16*)(ws + (16u << 20));   //  8 MB (B,H,DH,S)
    float*    rel_sm = (float*)   (ws + (24u << 20));   //  8 MB
    float*    tim_sm = (float*)   (ws + (32u << 20));   //  8 MB
    float*    rmax   = (float*)   (ws + (40u << 20));   // 256 KB
    float*    rsum   = (float*)   (ws + (40u << 20) + (512u << 10));

    float* out  = (float*)d_out;                        // (B,S,D)
    float* prob = out + (size_t)Bsz * S * D;            // (B,H,S,S)

    const dim3 blk(256);
    proj_kernel<<<dim3(32, 8), blk, 0, stream>>>(query, Wq, bq, q16, 0);
    proj_kernel<<<dim3(32, 8), blk, 0, stream>>>(key_,  Wk, bk, k16, 0);
    proj_kernel<<<dim3(32, 8), blk, 0, stream>>>(value, Wv, bv, vT16, 1);
    rel_time_kernel<<<Bsz * S, 512, 0, stream>>>(rel, ts, rel_sm, tim_sm);
    attn_scores_kernel<<<dim3(4, Bsz * H), blk, 0, stream>>>(q16, k16, prob, rmax, rsum);
    combine_kernel<<<1024, 256, 0, stream>>>(prob, rmax, rsum, rel_sm, tim_sm, l1, l2);
    out_gemm_kernel<<<dim3(4, Bsz * H), blk, 0, stream>>>(prob, vT16, out);
}
